// masked_multi_head_attention_28948079575364
// MI455X (gfx1250) — matrix operand order repaired and checked by reading, not compiled, not run on hardware
//
#include <hip/hip_runtime.h>
#include <hip/hip_bf16.h>

// ---------------------------------------------------------------------------
// Types for CDNA5 WMMA (gfx1250, wave32)
// ---------------------------------------------------------------------------
typedef __bf16 bf16;
typedef __attribute__((ext_vector_type(16))) __bf16 v16bf;
typedef __attribute__((ext_vector_type(8)))  __bf16 v8bf;
typedef __attribute__((ext_vector_type(8)))  float  v8f;

#define SEQ   4096
#define DIM   1024
#define HEADS 16
#define HS    64

#define CAT16(lo, hi) __builtin_shufflevector(lo, hi, 0,1,2,3,4,5,6,7,8,9,10,11,12,13,14,15)

__device__ __forceinline__ v8f wmma_bf16(v16bf a, v16bf b, v8f c) {
    // D = A(16x32 bf16) * B(32x16 bf16) + C(16x16 f32)
    return __builtin_amdgcn_wmma_f32_16x16x32_bf16(
        false, a, false, b, (short)0, c, false, false);
}

// ---------------------------------------------------------------------------
// CDNA5 async copy: one instruction moves 16B/lane global -> LDS (ASYNCcnt).
// LDS flat addresses truncate to addr[31:0], so the low half of a generic
// pointer to a __shared__ object is the LDS byte address.
// ---------------------------------------------------------------------------
__device__ __forceinline__ void async_copy_b128(const bf16* lds_dst, const bf16* gsrc) {
    unsigned lds_off = (unsigned)(uintptr_t)lds_dst;
    asm volatile("global_load_async_to_lds_b128 %0, %1, off"
                 :: "v"(lds_off), "v"(gsrc) : "memory");
}
__device__ __forceinline__ void wait_async_le1() {
    asm volatile("s_wait_asynccnt 0x1" ::: "memory");
}

// ---------------------------------------------------------------------------
// fp32 -> bf16 conversion
// ---------------------------------------------------------------------------
__global__ void cvt_f32_bf16(const float* __restrict__ in, bf16* __restrict__ out, int n) {
    int i = blockIdx.x * blockDim.x + threadIdx.x;
    int stride = gridDim.x * blockDim.x;
    for (; i < n; i += stride) out[i] = (bf16)in[i];
}

// ---------------------------------------------------------------------------
// GEMM:  out[s, o] = sum_d A[s, d] * W[o, d] + bias[o]
// Block = 8 waves, tile 256(M) x 64(N); each wave owns 32 M-rows.
// B panels (64x32) staged to LDS with global_load_async_to_lds_b128,
// triple buffered, ONE barrier per k-step, branch-free inner loop
// (tail over-reads stay inside the workspace by construction).
// All 4 B fragments are preloaded into distinct registers so the LDS
// latency is hidden by staged partial s_wait_dscnt under the WMMA stream.
// ---------------------------------------------------------------------------
__global__ __launch_bounds__(256) void gemm_xwT(
    const bf16* __restrict__ A, const bf16* __restrict__ W,
    const float* __restrict__ bias,
    bf16* __restrict__ outB, float* __restrict__ outF, int vTranspose)
{
    __shared__ alignas(128) bf16 Bsm[3][64 * 32];   // 3 x 4KB B panels
    const unsigned PANEL = 64 * 32;                  // elements

    const int lane = threadIdx.x & 31;
    const int wv   = threadIdx.x >> 5;
    const int bm   = blockIdx.x >> 4;
    const int bn   = blockIdx.x & 15;
    const int row0 = bm * 256 + wv * 32;
    const int col0 = bn * 64;

    const int m   = lane & 15;
    const int kbA = (lane < 16) ? 0 : 16;
    const int n   = lane & 15;
    const int khB = (lane < 16) ? 0 : 16;

    // async-copy mapping: wave wv copies rows [wv*8, wv*8+8): 8 waves = 4KB panel
    const int brow = wv * 8 + (lane >> 2);
    const int bseg = (lane & 3) * 8;
    const bf16* bsrc = W + (size_t)(col0 + brow) * DIM + bseg;
    bf16* bdst = &Bsm[0][brow * 32 + bseg];

    // prologue: stage k-steps 0 and 1 into panels 0 and 1
    async_copy_b128(bdst + 0 * PANEL, bsrc + 0);
    async_copy_b128(bdst + 1 * PANEL, bsrc + 32);

    const bf16* arow0 = A + (size_t)(row0 + m) * DIM + kbA;
    const bf16* arow1 = A + (size_t)(row0 + 16 + m) * DIM + kbA;

    v16bf a0, a1;
    {
        v8bf l0 = *(const v8bf*)(arow0), h0 = *(const v8bf*)(arow0 + 8);
        v8bf l1 = *(const v8bf*)(arow1), h1 = *(const v8bf*)(arow1 + 8);
        a0 = CAT16(l0, h0);
        a1 = CAT16(l1, h1);
    }

    v8f acc[2][4];
    for (int i = 0; i < 2; ++i)
        for (int t = 0; t < 4; ++t) acc[i][t] = v8f{};

    int cur = 0;
    for (int k0 = 0; k0 < DIM; k0 += 32) {
        // invariant: exactly 2 copies in flight; oldest (this k-step) done after le1.
        wait_async_le1();
        __syncthreads();

        // issue k-step +2 into the panel last consumed at the previous iteration
        int wrt = cur + 2; if (wrt >= 3) wrt -= 3;
        async_copy_b128(bdst + (unsigned)wrt * PANEL, bsrc + k0 + 64);

        // unconditional A prefetch for the next k-step (overlaps WMMAs)
        v16bf na0, na1;
        {
            const bf16* p0 = arow0 + k0 + 32;
            const bf16* p1 = arow1 + k0 + 32;
            v8bf l0 = *(const v8bf*)(p0), h0 = *(const v8bf*)(p0 + 8);
            v8bf l1 = *(const v8bf*)(p1), h1 = *(const v8bf*)(p1 + 8);
            na0 = CAT16(l0, h0);
            na1 = CAT16(l1, h1);
        }

        // preload ALL B fragments into distinct regs -> one ds clause,
        // staged partial dscnt waits instead of 4x full stalls
        const bf16* bp = &Bsm[cur][0];
        v16bf b0 = *(const v16bf*)(bp + ( 0 + n) * 32 + khB);
        v16bf b1 = *(const v16bf*)(bp + (16 + n) * 32 + khB);
        v16bf b2 = *(const v16bf*)(bp + (32 + n) * 32 + khB);
        v16bf b3 = *(const v16bf*)(bp + (48 + n) * 32 + khB);

        acc[0][0] = wmma_bf16(a0, b0, acc[0][0]);
        acc[1][0] = wmma_bf16(a1, b0, acc[1][0]);
        acc[0][1] = wmma_bf16(a0, b1, acc[0][1]);
        acc[1][1] = wmma_bf16(a1, b1, acc[1][1]);
        acc[0][2] = wmma_bf16(a0, b2, acc[0][2]);
        acc[1][2] = wmma_bf16(a1, b2, acc[1][2]);
        acc[0][3] = wmma_bf16(a0, b3, acc[0][3]);
        acc[1][3] = wmma_bf16(a1, b3, acc[1][3]);

        a0 = na0;
        a1 = na1;
        ++cur; if (cur >= 3) cur = 0;
    }

    // Epilogue: C/D layout -> lane holds col n; row = r + ((lane<16)?0:8)
    const int rofs = (lane < 16) ? 0 : 8;
    if (outF) {
        for (int t = 0; t < 4; ++t) {
            const int col = col0 + t * 16 + n;
            const float bv = bias[col];
            for (int i = 0; i < 2; ++i)
                for (int r = 0; r < 8; ++r)
                    outF[(size_t)(row0 + i * 16 + r + rofs) * DIM + col] = acc[i][t][r] + bv;
        }
    } else if (vTranspose) {
        // V stored [DIM, SEQ]: rows contiguous per lane -> packed 16B stores
        for (int t = 0; t < 4; ++t) {
            const int col = col0 + t * 16 + n;
            const float bv = bias[col];
            for (int i = 0; i < 2; ++i) {
                v8bf pk;
                for (int r = 0; r < 8; ++r) pk[r] = (bf16)(acc[i][t][r] + bv);
                *(v8bf*)(outB + (size_t)col * SEQ + row0 + i * 16 + rofs) = pk;
            }
        }
    } else {
        for (int t = 0; t < 4; ++t) {
            const int col = col0 + t * 16 + n;
            const float bv = bias[col];
            for (int i = 0; i < 2; ++i)
                for (int r = 0; r < 8; ++r)
                    outB[(size_t)(row0 + i * 16 + r + rofs) * DIM + col] = (bf16)(acc[i][t][r] + bv);
        }
    }
}

// ---------------------------------------------------------------------------
// Causal flash attention, one wave per (head, 16-query tile).  LDS-free:
// scores are computed TRANSPOSED (T = K * Q^T) so the D-layout of T
// (lane = query, VGPR = key) converts in-lane into the A-fragment of the
// P*V WMMA: ap[e<8]=p(T0 row e), ap[e>=8]=p(T1 row e-8).  Softmax stats are
// one scalar per lane (one query per lane); cross-half combine = shfl_xor(16).
// ---------------------------------------------------------------------------
__global__ __launch_bounds__(32) void attn_fwd(
    const bf16* __restrict__ Q, const bf16* __restrict__ K,
    const bf16* __restrict__ Vt, bf16* __restrict__ O)
{
    const int lane = threadIdx.x & 31;
    const int h  = blockIdx.x >> 8;
    const int qt = blockIdx.x & 255;
    const int q0 = qt * 16;

    const int nn   = lane & 15;
    const int half = (lane < 16) ? 0 : 1;
    const int kbA  = half ? 8 : 0;     // A-fragment k-base
    const int khB  = half ? 16 : 0;    // B-fragment k-base
    const int rofs = half ? 8 : 0;     // D-layout row offset

    // B-fragments of Q^T (fixed for the whole row of key chunks)
    v16bf bq0, bq1;
    {
        const bf16* p = Q + (size_t)(q0 + nn) * DIM + h * HS + kbA;
        bq0 = CAT16(*(const v8bf*)(p), *(const v8bf*)(p + 16));
        bq1 = CAT16(*(const v8bf*)(p + 32), *(const v8bf*)(p + 48));
    }

    // A-fragment of K rows [key0, key0+16) over d-chunk dch
    auto load_k_afrag = [&](int key0, int dch) -> v16bf {
        const bf16* p = K + (size_t)(key0 + nn) * DIM + h * HS + dch * 32 + kbA;
        v8bf lo = *(const v8bf*)(p);
        v8bf hi = *(const v8bf*)(p + 16);
        return CAT16(lo, hi);
    };

    v8f o[4] = {v8f{}, v8f{}, v8f{}, v8f{}};
    float m_run = -3.0e38f, l_run = 0.0f;
    const float scale = 0.125f;                     // 1/sqrt(64)
    const int query = q0 + nn;
    const int nChunks = (q0 + 16 + 31) >> 5;

    const bf16* vbase = Vt + (size_t)(h * HS + nn) * SEQ + kbA;

    // preload K A-fragments for chunk 0: ka[st][dch]
    v16bf ka[2][2];
    ka[0][0] = load_k_afrag(0, 0);  ka[0][1] = load_k_afrag(0, 1);
    ka[1][0] = load_k_afrag(16, 0); ka[1][1] = load_k_afrag(16, 1);

    for (int c = 0; c < nChunks; ++c) {
        const int kc0 = c * 32;

        // V fragments for this chunk (independent -> issue early)
        v16bf vf[4];
        for (int t = 0; t < 4; ++t)
            vf[t] = CAT16(*(const v8bf*)(vbase + (size_t)(t * 16) * SEQ + kc0), *(const v8bf*)(vbase + (size_t)(t * 16) * SEQ + kc0 + 16));

        // transposed scores: T(m=key, n=query), K-dim 64 = 2 WMMA per tile
        v8f t0 = v8f{}, t1 = v8f{};
        t0 = wmma_bf16(ka[0][0], bq0, t0);
        t0 = wmma_bf16(ka[0][1], bq1, t0);
        t1 = wmma_bf16(ka[1][0], bq0, t1);
        t1 = wmma_bf16(ka[1][1], bq1, t1);

        // unconditional prefetch of next chunk's K fragments (tail over-read ok)
        ka[0][0] = load_k_afrag(kc0 + 32, 0); ka[0][1] = load_k_afrag(kc0 + 32, 1);
        ka[1][0] = load_k_afrag(kc0 + 48, 0); ka[1][1] = load_k_afrag(kc0 + 48, 1);

        // mask + per-lane softmax (this lane owns one query column)
        float v0[8], v1[8];
        float tmax = -3.0e38f;
        for (int r = 0; r < 8; ++r) {
            float a = t0[r] * scale;
            float b = t1[r] * scale;
            if (kc0 + r + rofs      > query) a = -1.0e9f;
            if (kc0 + 16 + r + rofs > query) b = -1.0e9f;
            v0[r] = a; v1[r] = b;
            tmax = fmaxf(tmax, fmaxf(a, b));
        }
        tmax = fmaxf(tmax, __shfl_xor(tmax, 16));
        const float mnew = fmaxf(m_run, tmax);
        const float sc = __expf(m_run - mnew);

        v16bf ap;
        float psum = 0.0f;
        for (int r = 0; r < 8; ++r) {
            const float p0 = __expf(v0[r] - mnew);
            const float p1 = __expf(v1[r] - mnew);
            psum += p0 + p1;
            ap[r]     = (bf16)p0;
            ap[r + 8] = (bf16)p1;
        }
        psum += __shfl_xor(psum, 16);
        l_run = l_run * sc + psum;
        m_run = mnew;

        // rescale O rows (row r+rofs scale lives in lane r+rofs, equal per half)
        for (int r = 0; r < 8; ++r) {
            const float srow = __shfl(sc, r + rofs);
            o[0][r] *= srow; o[1][r] *= srow; o[2][r] *= srow; o[3][r] *= srow;
        }

        // P x V -> accumulate O (D layout: lane = d, VGPR = query row)
        for (int t = 0; t < 4; ++t)
            o[t] = wmma_bf16(ap, vf[t], o[t]);
    }

    // normalize + store (row-major [SEQ, DIM] bf16)
    const float linv = 1.0f / l_run;
    for (int r = 0; r < 8; ++r) {
        const float ir = __shfl(linv, r + rofs);
        const int row = q0 + r + rofs;
        for (int t = 0; t < 4; ++t) {
            const int col = h * HS + t * 16 + nn;
            O[(size_t)row * DIM + col] = (bf16)(o[t][r] * ir);
        }
    }
}

// ---------------------------------------------------------------------------
// Launch.  Workspace layout keeps every GEMM input followed by another
// region so branch-free tail over-reads stay inside the workspace:
//   [0,8) Xh  [8,16) Attn  [16,18) Wq [18,20) Wk [20,22) Wv [22,24) Wo
//   [24,32) Qb  [32,40) Kb  [40,48) Vtb            (MiB)
// ---------------------------------------------------------------------------
extern "C" void kernel_launch(void* const* d_in, const int* in_sizes, int n_in,
                              void* d_out, int out_size, void* d_ws, size_t ws_size,
                              hipStream_t stream) {
    const float* x  = (const float*)d_in[0];
    const float* wq = (const float*)d_in[1];
    const float* bq = (const float*)d_in[2];
    const float* wk = (const float*)d_in[3];
    const float* bk = (const float*)d_in[4];
    const float* wv = (const float*)d_in[5];
    const float* bv = (const float*)d_in[6];
    const float* wo = (const float*)d_in[7];
    const float* bo = (const float*)d_in[8];
    float* out = (float*)d_out;

    char* ws = (char*)d_ws;
    const size_t MB = 1024 * 1024;
    bf16* Xh   = (bf16*)(ws + 0 * MB);
    bf16* Attn = (bf16*)(ws + 8 * MB);
    bf16* Wqh  = (bf16*)(ws + 16 * MB);
    bf16* Wkh  = (bf16*)(ws + 18 * MB);
    bf16* Wvh  = (bf16*)(ws + 20 * MB);
    bf16* Woh  = (bf16*)(ws + 22 * MB);
    bf16* Qb   = (bf16*)(ws + 24 * MB);
    bf16* Kb   = (bf16*)(ws + 32 * MB);
    bf16* Vtb  = (bf16*)(ws + 40 * MB);

    // Stage 0: fp32 -> bf16
    cvt_f32_bf16<<<4096, 256, 0, stream>>>(x,  Xh,  SEQ * DIM);
    cvt_f32_bf16<<<1024, 256, 0, stream>>>(wq, Wqh, DIM * DIM);
    cvt_f32_bf16<<<1024, 256, 0, stream>>>(wk, Wkh, DIM * DIM);
    cvt_f32_bf16<<<1024, 256, 0, stream>>>(wv, Wvh, DIM * DIM);
    cvt_f32_bf16<<<1024, 256, 0, stream>>>(wo, Woh, DIM * DIM);

    // Stage 1: QKV projections.  (SEQ/256)*(DIM/64) = 256 blocks of 8 waves
    gemm_xwT<<<256, 256, 0, stream>>>(Xh, Wqh, bq, Qb,  nullptr, 0);
    gemm_xwT<<<256, 256, 0, stream>>>(Xh, Wkh, bk, Kb,  nullptr, 0);
    gemm_xwT<<<256, 256, 0, stream>>>(Xh, Wvh, bv, Vtb, nullptr, 1);   // V transposed

    // Stage 2: causal flash attention.  HEADS * (SEQ/16) = 4096 single-wave blocks
    attn_fwd<<<HEADS * (SEQ / 16), 32, 0, stream>>>(Qb, Kb, Vtb, Attn);

    // Stage 3: output projection -> fp32 d_out
    gemm_xwT<<<256, 256, 0, stream>>>(Attn, Woh, bo, nullptr, out, 0);
}
